// Tagger_78984448573921
// MI455X (gfx1250) — compile-verified, matrix operand order checked by reading, not run on hardware
//
#include <hip/hip_runtime.h>
#include <hip/hip_bf16.h>
#include <math.h>

// ---------------- dims ----------------
#define T_SEQ 8192
#define E_DIM 20
#define H1 30
#define H2 50
#define HL 300
#define NT 50

typedef __attribute__((ext_vector_type(16))) _Float16 v16h;
typedef __attribute__((ext_vector_type(8)))  _Float16 v8h;
typedef __attribute__((ext_vector_type(8)))  float    v8f;

// ---------------- workspace layout (byte offsets, all 256-aligned) ----------------
static constexpr size_t OFF_W1   = 0;            // 240 x 32 f16  (fw1_Wih | bw1_Wih), K pad 20->32
static constexpr size_t OFF_W2   = 15360;        // 400 x 64 f16  (fw2_Wih | bw2_Wih), K pad 60->64
static constexpr size_t OFF_L1W  = 66560;        // 320 x 128 f16 (lin1_W), 300->320 rows, 100->128 K
static constexpr size_t OFF_L2W  = 148480;       // 64 x 320 f16  (lin2_W), 50->64 rows, 300->320 K
static constexpr size_t OFF_L1A  = 189440;       // 8192 x 64 f16  l1 = [f1|b1|pad]
static constexpr size_t OFF_L2A  = 1238016;      // 8192 x 128 f16 l2 = [f2|b2|pad]
static constexpr size_t OFF_BIAS = 3335168;      // 1024 f32: [b1cat 240][b2cat 400][lin1_b 320][lin2_b 64]
static constexpr size_t ZERO_BYTES = 3339264;    // zero-fill everything above
static constexpr size_t OFF_E16  = 3339264;      // 8192 x 32 f16 gathered embeddings (K pad 20->32)
static constexpr size_t OFF_GX1  = 3863552;      // 8192 x 240 f32
static constexpr size_t OFF_GX2  = 11727872;     // 8192 x 400 f32
static constexpr size_t OFF_HMID = 24835072;     // 8192 x 320 f16 tanh(lin1) (K pad 300->320)

// ---------------- utility kernels ----------------
__global__ void zero_u32_kernel(uint32_t* p, size_t n) {
    size_t i = (size_t)blockIdx.x * blockDim.x + threadIdx.x;
    if (i < n) p[i] = 0u;
}

__global__ void gather_embed_kernel(const int* __restrict__ x,
                                    const float* __restrict__ emb,
                                    _Float16* __restrict__ e16) {
    int idx = blockIdx.x * blockDim.x + threadIdx.x;      // T*32 threads
    if (idx >= T_SEQ * 32) return;
    int t = idx >> 5, k = idx & 31;
    float v = (k < E_DIM) ? emb[(size_t)x[t] * E_DIM + k] : 0.0f;
    e16[idx] = (_Float16)v;
}

__global__ void pack_w16_kernel(_Float16* __restrict__ dst, int dstLd,
                                const float* __restrict__ src, int rows, int cols) {
    int idx = blockIdx.x * blockDim.x + threadIdx.x;
    if (idx >= rows * cols) return;
    int r = idx / cols, c = idx % cols;
    dst[(size_t)r * dstLd + c] = (_Float16)src[idx];
}

__global__ void pack_b32_kernel(float* __restrict__ dst, const float* __restrict__ src, int n) {
    int idx = blockIdx.x * blockDim.x + threadIdx.x;
    if (idx < n) dst[idx] = src[idx];
}

// ---------------- WMMA GEMM: out[MxN] = A[MxK] * B^T[NxK] + bias ----------------
// Fragment layout per ISA 7.12.2: lane<16 holds K{0-7,16-23}, lane>=16 holds K{8-15,24-31}.
// mode 0: f32 out = acc+bias ; mode 1: f16 out = tanh(acc+bias) ; mode 2: f32 out, col<nMax only
__global__ __launch_bounds__(32) void wmma_gemm_kernel(
        const _Float16* __restrict__ A, int ldA,
        const _Float16* __restrict__ B, int ldB,
        const float* __restrict__ bias,
        void* __restrict__ out, int ldOut,
        int kTiles, int mode, int nMax) {
    const int lane = threadIdx.x;
    const int n15  = lane & 15;
    const int hsel = lane >> 4;
    const _Float16* ap = A + (size_t)(blockIdx.x * 16 + n15) * ldA + hsel * 8;
    const _Float16* bp = B + (size_t)(blockIdx.y * 16 + n15) * ldB + hsel * 8;

    v8f acc = {0.f, 0.f, 0.f, 0.f, 0.f, 0.f, 0.f, 0.f};
    for (int kb = 0; kb < kTiles; ++kb) {
        v8h alo = *(const v8h*)(ap + kb * 32);
        v8h ahi = *(const v8h*)(ap + kb * 32 + 16);
        v8h blo = *(const v8h*)(bp + kb * 32);
        v8h bhi = *(const v8h*)(bp + kb * 32 + 16);
        v16h av, bv;
#pragma unroll
        for (int i = 0; i < 8; ++i) {
            av[i] = alo[i]; av[i + 8] = ahi[i];
            bv[i] = blo[i]; bv[i + 8] = bhi[i];
        }
        acc = __builtin_amdgcn_wmma_f32_16x16x32_f16(
            /*neg_a=*/false, av, /*neg_b=*/false, bv,
            /*c_mod=*/(short)0, acc, /*reuse_a=*/false, /*reuse_b=*/false);
    }

    const int   col = blockIdx.y * 16 + n15;
    const float bb  = bias[col];
#pragma unroll
    for (int r = 0; r < 8; ++r) {
        size_t row = (size_t)(blockIdx.x * 16 + r + hsel * 8);
        float  v   = acc[r] + bb;
        if (mode == 0) {
            ((float*)out)[row * ldOut + col] = v;
        } else if (mode == 1) {
            ((_Float16*)out)[row * ldOut + col] = (_Float16)tanhf(v);
        } else {
            if (col < nMax) ((float*)out)[row * ldOut + col] = v;
        }
    }
}

// ---------------- bidirectional LSTM scan (register-resident Whh) ----------------
// block 0 = forward, block 1 = backward (reads gx[T-1-t], writes row t -> iteration order,
// matching the reference's un-flipped concat). Bias already folded into gx.
// One thread per h-index computes all 4 of its gate dots (no cross-thread gate handoff).
// h exchanged via ping-pong LDS buffer -> ONE barrier per step. gx software-pipelined in
// registers; global_prefetch issued ~16 steps ahead.
template<int H, int THREADS>
__global__ __launch_bounds__(THREADS) void lstm_scan_reg_kernel(
        const float* __restrict__ gx, int gxLd,
        const float* __restrict__ WhhF, const float* __restrict__ WhhB,
        const float* __restrict__ h0f, const float* __restrict__ c0f,
        const float* __restrict__ h0b, const float* __restrict__ c0b,
        _Float16* __restrict__ out, int outLd) {
    __shared__ float hbuf[2][64];

    const bool   bw  = (blockIdx.x == 1);
    const float* Whh = bw ? WhhB : WhhF;
    const float* h0  = bw ? h0b : h0f;
    const float* c0  = bw ? c0b : c0f;
    const int    gxCol  = bw ? 4 * H : 0;
    const int    outCol = bw ? H : 0;
    const int    j   = threadIdx.x;
    const bool   act = (j < H);
    const int    jc  = act ? j : (H - 1);   // clamped index: keeps all loads in-bounds

    // recurrent weights in registers: w[g][k] = Whh[(g*H + j)*H + k]
    float w[4][H];
#pragma unroll
    for (int g = 0; g < 4; ++g)
#pragma unroll
        for (int k = 0; k < H; ++k)
            w[g][k] = Whh[(size_t)(g * H + jc) * H + k];

    float c = c0[jc];
    if (act) hbuf[0][j] = h0[j];
    __syncthreads();

    // preload gx for t = 0
    float gcur[4];
    {
        const int tg0 = bw ? (T_SEQ - 1) : 0;
#pragma unroll
        for (int g = 0; g < 4; ++g)
            gcur[g] = gx[(size_t)tg0 * gxLd + gxCol + g * H + jc];
    }

    for (int t = 0; t < T_SEQ; ++t) {
        const int cur = t & 1;

        // issue next step's gx loads now (off the critical chain)
        float gnext[4];
        const int tn = bw ? (T_SEQ - 2 - t) : (t + 1);
        if (t + 1 < T_SEQ) {
#pragma unroll
            for (int g = 0; g < 4; ++g)
                gnext[g] = gx[(size_t)tn * gxLd + gxCol + g * H + jc];
        } else {
#pragma unroll
            for (int g = 0; g < 4; ++g) gnext[g] = 0.0f;
        }
        // L2 prefetch ~16 steps ahead
        const int tp = bw ? (T_SEQ - 1 - t - 16) : (t + 16);
        if (tp >= 0 && tp < T_SEQ)
            __builtin_prefetch(gx + (size_t)tp * gxLd + gxCol + jc, 0, 1);

        float a0 = gcur[0], a1 = gcur[1], a2 = gcur[2], a3 = gcur[3];
#pragma unroll
        for (int k = 0; k < H; ++k) {
            const float hk = hbuf[cur][k];     // uniform broadcast read
            a0 += w[0][k] * hk;
            a1 += w[1][k] * hk;
            a2 += w[2][k] * hk;
            a3 += w[3][k] * hk;
        }
        const float i_ = 1.0f / (1.0f + __expf(-a0));
        const float f_ = 1.0f / (1.0f + __expf(-a1));
        const float g_ = tanhf(a2);
        const float o_ = 1.0f / (1.0f + __expf(-a3));
        c = f_ * c + i_ * g_;
        const float h = o_ * tanhf(c);
        if (act) {
            hbuf[cur ^ 1][j] = h;
            out[(size_t)t * outLd + outCol + j] = (_Float16)h;
        }
        __syncthreads();   // single barrier: covers hbuf write->read and read->overwrite
#pragma unroll
        for (int g = 0; g < 4; ++g) gcur[g] = gnext[g];
    }
}

// ---------------- host launcher ----------------
extern "C" void kernel_launch(void* const* d_in, const int* in_sizes, int n_in,
                              void* d_out, int out_size, void* d_ws, size_t ws_size,
                              hipStream_t stream) {
    (void)in_sizes; (void)n_in; (void)out_size; (void)ws_size;

    const int*   x        = (const int*)d_in[0];
    const float* emb      = (const float*)d_in[1];
    const float* fw1_Wih  = (const float*)d_in[2];
    const float* fw1_Whh  = (const float*)d_in[3];
    const float* fw1_b    = (const float*)d_in[4];
    const float* fw1_h0   = (const float*)d_in[5];
    const float* fw1_c0   = (const float*)d_in[6];
    const float* bw1_Wih  = (const float*)d_in[7];
    const float* bw1_Whh  = (const float*)d_in[8];
    const float* bw1_b    = (const float*)d_in[9];
    const float* bw1_h0   = (const float*)d_in[10];
    const float* bw1_c0   = (const float*)d_in[11];
    const float* fw2_Wih  = (const float*)d_in[12];
    const float* fw2_Whh  = (const float*)d_in[13];
    const float* fw2_b    = (const float*)d_in[14];
    const float* fw2_h0   = (const float*)d_in[15];
    const float* fw2_c0   = (const float*)d_in[16];
    const float* bw2_Wih  = (const float*)d_in[17];
    const float* bw2_Whh  = (const float*)d_in[18];
    const float* bw2_b    = (const float*)d_in[19];
    const float* bw2_h0   = (const float*)d_in[20];
    const float* bw2_c0   = (const float*)d_in[21];
    const float* lin1_W   = (const float*)d_in[22];
    const float* lin1_b   = (const float*)d_in[23];
    const float* lin2_W   = (const float*)d_in[24];
    const float* lin2_b   = (const float*)d_in[25];

    char* ws = (char*)d_ws;
    _Float16* W1   = (_Float16*)(ws + OFF_W1);
    _Float16* W2   = (_Float16*)(ws + OFF_W2);
    _Float16* L1W  = (_Float16*)(ws + OFF_L1W);
    _Float16* L2W  = (_Float16*)(ws + OFF_L2W);
    _Float16* l1a  = (_Float16*)(ws + OFF_L1A);
    _Float16* l2a  = (_Float16*)(ws + OFF_L2A);
    float*    bias = (float*)   (ws + OFF_BIAS);
    _Float16* e16  = (_Float16*)(ws + OFF_E16);
    float*    gx1  = (float*)   (ws + OFF_GX1);
    float*    gx2  = (float*)   (ws + OFF_GX2);
    _Float16* hmid = (_Float16*)(ws + OFF_HMID);

    // 1) zero-fill pad regions
    {
        size_t n32 = ZERO_BYTES / 4;
        zero_u32_kernel<<<(unsigned)((n32 + 255) / 256), 256, 0, stream>>>((uint32_t*)ws, n32);
    }
    // 2) gather embeddings -> f16, K padded to 32
    gather_embed_kernel<<<(T_SEQ * 32) / 256, 256, 0, stream>>>(x, emb, e16);
    // 3) pack weights to f16 arenas
    auto packW = [&](_Float16* dst, int ld, const float* src, int r, int c) {
        pack_w16_kernel<<<(r * c + 255) / 256, 256, 0, stream>>>(dst, ld, src, r, c);
    };
    packW(W1,                 32, fw1_Wih, 120, 20);
    packW(W1 + 120 * 32,      32, bw1_Wih, 120, 20);
    packW(W2,                 64, fw2_Wih, 200, 60);
    packW(W2 + 200 * 64,      64, bw2_Wih, 200, 60);
    packW(L1W,               128, lin1_W,  300, 100);
    packW(L2W,               320, lin2_W,   50, 300);
    // 4) pack biases (pads already zero)
    auto packB = [&](float* dst, const float* src, int n) {
        pack_b32_kernel<<<(n + 255) / 256, 256, 0, stream>>>(dst, src, n);
    };
    packB(bias + 0,   fw1_b, 120);
    packB(bias + 120, bw1_b, 120);
    packB(bias + 240, fw2_b, 200);
    packB(bias + 440, bw2_b, 200);
    packB(bias + 640, lin1_b, 300);
    packB(bias + 960, lin2_b, 50);

    // 5) gx1[T x 240] = e16 @ [fw1|bw1 Wih]^T + b   (K = 32)
    wmma_gemm_kernel<<<dim3(T_SEQ / 16, 240 / 16), 32, 0, stream>>>(
        e16, 32, W1, 32, bias + 0, gx1, 240, 1, 0, 0);
    // 6) layer-1 scans (fw + bw concurrently), H = 30, one wave each
    lstm_scan_reg_kernel<H1, 32><<<2, 32, 0, stream>>>(
        gx1, 240, fw1_Whh, bw1_Whh, fw1_h0, fw1_c0, bw1_h0, bw1_c0, l1a, 64);
    // 7) gx2[T x 400] = l1 @ [fw2|bw2 Wih]^T + b   (K = 64)
    wmma_gemm_kernel<<<dim3(T_SEQ / 16, 400 / 16), 32, 0, stream>>>(
        l1a, 64, W2, 64, bias + 240, gx2, 400, 2, 0, 0);
    // 8) layer-2 scans, H = 50, two waves each
    lstm_scan_reg_kernel<H2, 64><<<2, 64, 0, stream>>>(
        gx2, 400, fw2_Whh, bw2_Whh, fw2_h0, fw2_c0, bw2_h0, bw2_c0, l2a, 128);
    // 9) hmid[T x 320] = tanh(l2 @ lin1_W^T + b)   (K = 128)
    wmma_gemm_kernel<<<dim3(T_SEQ / 16, 320 / 16), 32, 0, stream>>>(
        l2a, 128, L1W, 128, bias + 640, hmid, 320, 4, 1, 0);
    // 10) out[T x 50] = hmid @ lin2_W^T + b        (K = 320, mask cols >= 50)
    wmma_gemm_kernel<<<dim3(T_SEQ / 16, 64 / 16), 32, 0, stream>>>(
        hmid, 320, L2W, 320, bias + 960, (float*)d_out, NT, 10, 2, NT);
}
